// MHA_66443144069648
// MI455X (gfx1250) — matrix operand order repaired and checked by reading, not compiled, not run on hardware
//
#include <hip/hip_runtime.h>
#include <hip/hip_bf16.h>

// ---------------------------------------------------------------------------
// Problem constants (from reference)
// ---------------------------------------------------------------------------
#define S_LEN 2048
#define DIM   4096
#define NH    32
#define NKV   8
#define HD    128          // DIM / NH
#define KVDIM 1024         // NKV * HD

// ---------------------------------------------------------------------------
// CDNA5 WMMA / TDM types
// ---------------------------------------------------------------------------
typedef __attribute__((ext_vector_type(16))) __bf16   v16bf;
typedef __attribute__((ext_vector_type(8)))  float    v8f;
typedef __attribute__((ext_vector_type(4)))  unsigned v4u;
typedef __attribute__((ext_vector_type(8)))  int      v8i;
typedef __attribute__((ext_vector_type(4)))  int      v4i;

#if __has_builtin(__builtin_amdgcn_tensor_load_to_lds)
#define HAVE_TDM 1
#else
#define HAVE_TDM 0
#endif

union AFrag {                      // 16 bf16 values per lane = 8 VGPRs
    v16bf v;
    unsigned short u[16];
};

__device__ __forceinline__ v8f wmma_bf16(v16bf a, v16bf b, v8f c) {
    return __builtin_amdgcn_wmma_f32_16x16x32_bf16(
        /*neg_a=*/false, a, /*neg_b=*/false, b,
        /*c_mod=*/(short)0, c, /*reuse_a=*/false, /*reuse_b=*/false);
}

__device__ __forceinline__ unsigned short f2bf(float f) {
    unsigned int u = __float_as_uint(f);
    unsigned int r = u + 0x7FFFu + ((u >> 16) & 1u);   // round-to-nearest-even
    return (unsigned short)(r >> 16);
}

#if HAVE_TDM
// ---------------------------------------------------------------------------
// Tensor Data Mover: 2D tile load (bf16 elements) global -> LDS.
//   lds_off          : LDS byte offset of destination tile (row-major)
//   gaddr            : global address of tile start
//   rows, cols       : tile_dim1 (Y), tile_dim0 (X) in elements
//   tensor_w, tensor_h, row_stride : tensor dims / dim0 stride in elements
// D# layout per CDNA5 ISA 8.3/8.4 (group0 128b, group1 256b).
// This toolchain exposes the 6-arg builtin:
//   (v4u g0, v8i g1, v4i g2, v4i g3, v8i g4, i32 cpol)
// ---------------------------------------------------------------------------
__device__ __forceinline__ void tdm_load_tile_bf16(
        unsigned lds_off, const void* gaddr,
        unsigned rows, unsigned cols,
        unsigned tensor_w, unsigned tensor_h, unsigned row_stride) {
    unsigned long long ga = (unsigned long long)gaddr;
    v4u g0;
    g0.x = 1u;                                       // count=1, is_restore=0, no gather
    g0.y = lds_off;                                  // lds_addr[31:0]
    g0.z = (unsigned)(ga & 0xFFFFFFFFu);             // global_addr[31:0]
    g0.w = (unsigned)((ga >> 32) & 0x01FFFFFFu)      // global_addr[56:32]
           | 0x80000000u;                            // type=2 ("image") at [127:126]
    v8i g1;
    g1[0] = 0x00010000;                              // data_size=1 (2 bytes), mask=0
    g1[1] = (int)((tensor_w & 0xFFFFu) << 16);       // tensor_dim0[15:0] (atomic addr=0)
    g1[2] = (int)((tensor_w >> 16) | ((tensor_h & 0xFFFFu) << 16)); // dim0 hi | dim1 lo
    g1[3] = (int)((tensor_h >> 16) | (cols << 16));  // dim1 hi | tile_dim0
    g1[4] = (int)rows;                               // tile_dim1 (tile_dim2=0)
    g1[5] = (int)row_stride;                         // tensor_dim0_stride[31:0]
    g1[6] = 0;                                       // stride0 hi | stride1 lo
    g1[7] = 0;                                       // tensor_dim1_stride hi
    v4i zero4 = {0, 0, 0, 0};                        // groups 2/3 unused (<=2D)
    v8i zero8 = {0, 0, 0, 0, 0, 0, 0, 0};
    __builtin_amdgcn_tensor_load_to_lds(g0, g1, zero4, zero4, zero8, 0);
}
#endif

// ---------------------------------------------------------------------------
// fp32 -> bf16 cast
// ---------------------------------------------------------------------------
__global__ void cast_f32_bf16(const float* __restrict__ in,
                              unsigned short* __restrict__ out, long long n) {
    long long i = (long long)blockIdx.x * blockDim.x + threadIdx.x;
    if (i < n) out[i] = f2bf(in[i]);
}

// ---------------------------------------------------------------------------
// RoPE (interleaved-pair convention of the reference) + cast to bf16
// ---------------------------------------------------------------------------
__global__ void rope_cast(const float* __restrict__ in,
                          unsigned short* __restrict__ out, int nh) {
    int idx = blockIdx.x * blockDim.x + threadIdx.x;
    int total = S_LEN * nh * (HD / 2);
    if (idx >= total) return;
    int i = idx % (HD / 2);
    int h = (idx / (HD / 2)) % nh;
    int s = idx / ((HD / 2) * nh);
    size_t base = (size_t)s * (nh * HD) + (size_t)h * HD + 2 * i;
    float e0 = in[base];
    float e1 = in[base + 1];
    float fr = __expf(-(2.0f * (float)i / (float)HD) * 13.122363377403613f); // ln(5e5)
    float ang = (float)s * fr;
    float sn, cs;
    __sincosf(ang, &sn, &cs);
    out[base]     = f2bf(e0 * cs - e1 * sn);
    out[base + 1] = f2bf(e0 * sn + e1 * cs);
}

// ---------------------------------------------------------------------------
// bf16 WMMA GEMM:  C[M,N] (f32) = A[M,K] (bf16, row-major) * B[N,K]^T (bf16)
// Block tile 128(M) x 64(N), 256 threads = 8 waves, K stepped by 32.
// Tiles staged via Tensor Data Mover (TDM) when available.
// ---------------------------------------------------------------------------
__global__ __launch_bounds__(256) void gemm_bf16_nt(
        const unsigned short* __restrict__ A,
        const unsigned short* __restrict__ B,
        float* __restrict__ C, int M, int N, int K) {
    __shared__ __align__(16) unsigned short As[128][32];
    __shared__ __align__(16) unsigned short Bs[64][32];

    int t    = threadIdx.x;
    int lane = t & 31;
    int wave = t >> 5;
    int wm   = wave & 3;       // 0..3 -> 32-row group
    int wn   = wave >> 2;      // 0..1 -> 32-col group
    int half = lane >> 4;      // 0/1
    int l16  = lane & 15;

    int m0 = blockIdx.y * 128;
    int n0 = blockIdx.x * 64;

    const v8f zero = {0.f, 0.f, 0.f, 0.f, 0.f, 0.f, 0.f, 0.f};
    v8f acc[2][2];
    acc[0][0] = zero; acc[0][1] = zero; acc[1][0] = zero; acc[1][1] = zero;

#if HAVE_TDM
    unsigned lds_a = (unsigned)(unsigned long long)&As[0][0];
    unsigned lds_b = (unsigned)(unsigned long long)&Bs[0][0];
#endif

    for (int k0 = 0; k0 < K; k0 += 32) {
#if HAVE_TDM
        // --- TDM DMA: A tile 128x32, B tile 64x32, both row-major ---
        if (wave == 0) {
            tdm_load_tile_bf16(lds_a, A + (size_t)m0 * K + k0,
                               128, 32, (unsigned)K, (unsigned)M, (unsigned)K);
            tdm_load_tile_bf16(lds_b, B + (size_t)n0 * K + k0,
                               64, 32, (unsigned)K, (unsigned)N, (unsigned)K);
            __builtin_amdgcn_s_wait_tensorcnt(0);
        }
        __syncthreads();
#else
        // --- fallback: cooperative vector staging ---
        {
            int r = t >> 1;
            int c = (t & 1) * 16;
            const unsigned short* ap = A + (size_t)(m0 + r) * K + k0 + c;
            if (k0 + 32 < K) __builtin_prefetch(ap + 32, 0, 3);
            const uint4* src = (const uint4*)ap;
            uint4* dst = (uint4*)&As[r][c];
            dst[0] = src[0];
            dst[1] = src[1];
        }
        {
            int r = t >> 2;
            int c = (t & 3) * 8;
            const unsigned short* bp = B + (size_t)(n0 + r) * K + k0 + c;
            *(uint4*)&Bs[r][c] = *(const uint4*)bp;
        }
        __syncthreads();
#endif

        // --- build fragments and multiply ---
        AFrag af[2], bf[2];
        for (int mt = 0; mt < 2; ++mt) {
            int row = wm * 32 + mt * 16 + l16;
            #pragma unroll
            for (int j = 0; j < 8; ++j) {
                af[mt].u[j]     = As[row][half * 8 + j];        // K = 0..15
                af[mt].u[8 + j] = As[row][16 + half * 8 + j];   // K = 16..31
            }
        }
        for (int nt = 0; nt < 2; ++nt) {
            int cb = wn * 32 + nt * 16;
            #pragma unroll
            for (int n = 0; n < 16; ++n)
                bf[nt].u[n] = Bs[cb + l16][(n & 7) + half * 8 + (n >> 3) * 16];   // lane = output col, elem = K (table order)
        }
        #pragma unroll
        for (int mt = 0; mt < 2; ++mt)
            #pragma unroll
            for (int nt = 0; nt < 2; ++nt)
                acc[mt][nt] = wmma_bf16(af[mt].v, bf[nt].v, acc[mt][nt]);
        __syncthreads();
    }

    // --- write C (f32), C-fragment layout: VGPR e -> M = e (+8 for hi half) ---
    for (int mt = 0; mt < 2; ++mt)
        for (int nt = 0; nt < 2; ++nt) {
            int rowb = m0 + wm * 32 + mt * 16 + half * 8;
            int col  = n0 + wn * 32 + nt * 16 + l16;
            #pragma unroll
            for (int e = 0; e < 8; ++e)
                C[(size_t)(rowb + e) * N + col] = acc[mt][nt][e];
        }
}

// ---------------------------------------------------------------------------
// Flash attention (causal, GQA 32 heads sharing 8 KV heads)
//   Qb:  bf16 (S, NH,  HD)   after RoPE
//   Kb:  bf16 (S, NKV, HD)   after RoPE
//   Vb:  bf16 (S, NKV, HD)
//   Ctx: bf16 (S, NH*HD)
// Block = 128 threads = 4 independent waves; each wave owns a 16-row q tile.
// ---------------------------------------------------------------------------
__global__ __launch_bounds__(128) void flash_attn(
        const unsigned short* __restrict__ Qb,
        const unsigned short* __restrict__ Kb,
        const unsigned short* __restrict__ Vb,
        unsigned short* __restrict__ Ctx) {
    __shared__ __align__(16) unsigned short Ps[4][16][32];

    int lane  = threadIdx.x & 31;
    int wave  = threadIdx.x >> 5;
    int half  = lane >> 4;
    int l16   = lane & 15;
    int h     = blockIdx.x;
    int qbase = blockIdx.y * 64 + wave * 16;
    int kvh   = h >> 2;                  // NH/NKV = 4
    const int qstride = NH * HD;         // 4096
    const int kstride = NKV * HD;        // 1024
    const float scale = 0.08838834764831845f;  // 1/sqrt(128)

    // --- Q fragments: 4 chunks of K=32 along head dim ---
    AFrag qf[4];
    {
        const unsigned short* qrow =
            Qb + (size_t)(qbase + l16) * qstride + (size_t)h * HD;
        for (int c = 0; c < 4; ++c)
            #pragma unroll
            for (int j = 0; j < 8; ++j) {
                qf[c].u[j]     = qrow[c * 32 + half * 8 + j];
                qf[c].u[8 + j] = qrow[c * 32 + 16 + half * 8 + j];
            }
    }

    const v8f zero = {0.f, 0.f, 0.f, 0.f, 0.f, 0.f, 0.f, 0.f};
    v8f o[8];
    #pragma unroll
    for (int j = 0; j < 8; ++j) o[j] = zero;
    float mrow[8], lrow[8];
    #pragma unroll
    for (int e = 0; e < 8; ++e) { mrow[e] = -3.0e38f; lrow[e] = 0.f; }

    int kend = qbase + 16;               // causal upper bound for this q tile
    for (int kb = 0; kb < kend; kb += 32) {
        // ---- S = Q @ K^T : 16x32 as two 16x16 WMMA results ----
        v8f s[2];
        s[0] = zero; s[1] = zero;
        for (int nh2 = 0; nh2 < 2; ++nh2) {
            for (int c = 0; c < 4; ++c) {
                AFrag bfr;                                 // lane = key col n, elem = d (table order)
                const unsigned short* kp =
                    Kb + (size_t)(kb + nh2 * 16 + l16) * kstride +
                    (size_t)kvh * HD + c * 32;
                #pragma unroll
                for (int n = 0; n < 16; ++n)
                    bfr.u[n] = kp[(n & 7) + half * 8 + (n >> 3) * 16];    // B[d][n] = K[kb+n][d]
                s[nh2] = wmma_bf16(qf[c].v, bfr.v, s[nh2]);
            }
        }

        // ---- scale + causal mask ----
        for (int nh2 = 0; nh2 < 2; ++nh2)
            #pragma unroll
            for (int e = 0; e < 8; ++e) {
                int row = qbase + e + half * 8;
                int col = kb + nh2 * 16 + l16;
                float v = s[nh2][e] * scale;
                if (col > row) v = -1.0e9f;
                s[nh2][e] = v;
            }

        // ---- online softmax (row = (e, lane half); cols across 16 lanes) ----
        #pragma unroll
        for (int e = 0; e < 8; ++e) {
            float mx = fmaxf(s[0][e], s[1][e]);
            for (int off = 8; off; off >>= 1)
                mx = fmaxf(mx, __shfl_xor(mx, off, 32));
            float mnew = fmaxf(mrow[e], mx);
            float corr = __expf(mrow[e] - mnew);
            float p0 = __expf(s[0][e] - mnew);
            float p1 = __expf(s[1][e] - mnew);
            float rs = p0 + p1;
            for (int off = 8; off; off >>= 1)
                rs += __shfl_xor(rs, off, 32);
            lrow[e] = lrow[e] * corr + rs;
            mrow[e] = mnew;
            s[0][e] = p0;
            s[1][e] = p1;
            #pragma unroll
            for (int j = 0; j < 8; ++j) o[j][e] *= corr;   // rescale O row
        }

        // ---- stage P (16x32 bf16) through LDS: C-layout -> A-layout ----
        for (int nh2 = 0; nh2 < 2; ++nh2)
            #pragma unroll
            for (int e = 0; e < 8; ++e)
                Ps[wave][e + half * 8][nh2 * 16 + l16] = f2bf(s[nh2][e]);
        asm volatile("s_wait_dscnt 0" ::: "memory");       // per-wave LDS RAW fence

        AFrag pf;
        #pragma unroll
        for (int j = 0; j < 8; ++j) {
            pf.u[j]     = Ps[wave][l16][half * 8 + j];
            pf.u[8 + j] = Ps[wave][l16][16 + half * 8 + j];
        }

        // ---- O += P @ V : 8 WMMA over the 8 x 16-col chunks of HD ----
        for (int j = 0; j < 8; ++j) {
            AFrag vf;                                      // lane = col n, elem = k row (table order)
            const unsigned short* vp =
                Vb + (size_t)kb * kstride + (size_t)kvh * HD + j * 16 + l16;
            #pragma unroll
            for (int n = 0; n < 16; ++n) vf.u[n] = vp[(size_t)((n & 7) + half * 8 + (n >> 3) * 16) * kstride];
            o[j] = wmma_bf16(pf.v, vf.v, o[j]);
        }
    }

    // ---- normalize and write context (bf16) ----
    #pragma unroll
    for (int e = 0; e < 8; ++e) {
        float inv = 1.0f / lrow[e];
        int row = qbase + e + half * 8;
        #pragma unroll
        for (int j = 0; j < 8; ++j)
            Ctx[(size_t)row * (NH * HD) + (size_t)h * HD + j * 16 + l16] =
                f2bf(o[j][e] * inv);
    }
}

// ---------------------------------------------------------------------------
// Host-side orchestration
// ---------------------------------------------------------------------------
extern "C" void kernel_launch(void* const* d_in, const int* in_sizes, int n_in,
                              void* d_out, int out_size, void* d_ws, size_t ws_size,
                              hipStream_t stream) {
    const float* x  = (const float*)d_in[0];
    // d_in[1] = mask (unused: causal mask computed inline)
    const float* wq = (const float*)d_in[2];
    const float* wk = (const float*)d_in[3];
    const float* wv = (const float*)d_in[4];
    const float* wo = (const float*)d_in[5];
    float* out = (float*)d_out;

    // ---- workspace layout (bytes) ----
    char* ws = (char*)d_ws;
    const size_t SZ_XB  = (size_t)S_LEN * DIM   * 2;
    const size_t SZ_WQB = (size_t)DIM   * DIM   * 2;
    const size_t SZ_WKB = (size_t)KVDIM * DIM   * 2;
    const size_t SZ_WVB = (size_t)KVDIM * DIM   * 2;
    const size_t SZ_WOB = (size_t)DIM   * DIM   * 2;
    const size_t SZ_QF  = (size_t)S_LEN * DIM   * 4;
    const size_t SZ_KF  = (size_t)S_LEN * KVDIM * 4;
    const size_t SZ_VF  = (size_t)S_LEN * KVDIM * 4;
    const size_t SZ_QB  = (size_t)S_LEN * DIM   * 2;
    const size_t SZ_KB  = (size_t)S_LEN * KVDIM * 2;
    const size_t SZ_VB  = (size_t)S_LEN * KVDIM * 2;
    size_t off = 0;
    unsigned short* Xb  = (unsigned short*)(ws + off); off += SZ_XB;
    unsigned short* Wqb = (unsigned short*)(ws + off); off += SZ_WQB;
    unsigned short* Wkb = (unsigned short*)(ws + off); off += SZ_WKB;
    unsigned short* Wvb = (unsigned short*)(ws + off); off += SZ_WVB;
    unsigned short* Wob = (unsigned short*)(ws + off); off += SZ_WOB;
    float*          Qf  = (float*)(ws + off);          off += SZ_QF;
    float*          Kf  = (float*)(ws + off);          off += SZ_KF;
    float*          Vf  = (float*)(ws + off);          off += SZ_VF;
    unsigned short* Qb  = (unsigned short*)(ws + off); off += SZ_QB;
    unsigned short* Kb  = (unsigned short*)(ws + off); off += SZ_KB;
    unsigned short* Vb  = (unsigned short*)(ws + off); off += SZ_VB;
    unsigned short* Ctx = (unsigned short*)(ws + off);

    // ---- 1) cast inputs to bf16 ----
    auto cast = [&](const float* src, unsigned short* dst, long long n) {
        int blocks = (int)((n + 255) / 256);
        cast_f32_bf16<<<blocks, 256, 0, stream>>>(src, dst, n);
    };
    cast(x,  Xb,  (long long)S_LEN * DIM);
    cast(wq, Wqb, (long long)DIM   * DIM);
    cast(wk, Wkb, (long long)KVDIM * DIM);
    cast(wv, Wvb, (long long)KVDIM * DIM);
    cast(wo, Wob, (long long)DIM   * DIM);

    // ---- 2) projections: X @ W^T via WMMA ----
    gemm_bf16_nt<<<dim3(DIM / 64,   S_LEN / 128), 256, 0, stream>>>(
        Xb, Wqb, Qf, S_LEN, DIM,   DIM);
    gemm_bf16_nt<<<dim3(KVDIM / 64, S_LEN / 128), 256, 0, stream>>>(
        Xb, Wkb, Kf, S_LEN, KVDIM, DIM);
    gemm_bf16_nt<<<dim3(KVDIM / 64, S_LEN / 128), 256, 0, stream>>>(
        Xb, Wvb, Vf, S_LEN, KVDIM, DIM);

    // ---- 3) RoPE(Q,K) + cast, cast V ----
    {
        int nq = S_LEN * NH  * (HD / 2);
        int nk = S_LEN * NKV * (HD / 2);
        rope_cast<<<(nq + 255) / 256, 256, 0, stream>>>(Qf, Qb, NH);
        rope_cast<<<(nk + 255) / 256, 256, 0, stream>>>(Kf, Kb, NKV);
        cast(Vf, Vb, (long long)S_LEN * KVDIM);
    }

    // ---- 4) flash attention ----
    flash_attn<<<dim3(NH, S_LEN / 64), 128, 0, stream>>>(Qb, Kb, Vb, Ctx);

    // ---- 5) output projection: Ctx @ Wo^T -> out (f32) ----
    gemm_bf16_nt<<<dim3(DIM / 64, S_LEN / 128), 256, 0, stream>>>(
        Ctx, Wob, out, S_LEN, DIM, DIM);
}
